// ShuffleRowAttention_32452772889011
// MI455X (gfx1250) — matrix operand order repaired and checked by reading, not compiled, not run on hardware
//
#include <hip/hip_runtime.h>

typedef __attribute__((ext_vector_type(16))) __bf16 v16bf;
typedef __attribute__((ext_vector_type(4)))  __bf16 v4bf;
typedef __attribute__((ext_vector_type(2)))  __bf16 v2bf;
typedef __attribute__((ext_vector_type(8)))  float  v8f;
typedef __attribute__((ext_vector_type(8)))  unsigned int v8u;
typedef unsigned short u16;

constexpr int   CB     = 16;
constexpr int   CN     = 1024;
constexpr int   CDIM   = 768;
constexpr int   CH     = 12;
constexpr int   CDH    = 64;
constexpr float CSCALE = 0.125f;   // 64^-0.5

// ---------------------------------------------------------------------------
// CDNA5 helpers
// ---------------------------------------------------------------------------
__device__ __forceinline__ v8f wmma_bf16(v16bf a, v16bf b, v8f c) {
  return __builtin_amdgcn_wmma_f32_16x16x32_bf16(
      /*neg_a=*/false, a, /*neg_b=*/false, b,
      /*c_mod=*/(short)0, c, /*reuse_a=*/false, /*reuse_b=*/false);
}

// Low 32 bits of a generic pointer to __shared__ = LDS byte offset
// (FLAT->LDS aperture mapping keeps the low 32 address bits).
__device__ __forceinline__ unsigned lds_off(const void* p) {
  return (unsigned)(unsigned long long)p;
}

// Async global->LDS 16B copy per lane (GLOBAL_LOAD_ASYNC_TO_LDS_B128,
// tracked by ASYNCcnt; no VGPR data movement).
__device__ __forceinline__ void async_copy16(unsigned lds_dst,
                                             const void* gsrc) {
  asm volatile("global_load_async_to_lds_b128 %0, %1, off"
               :
               : "v"(lds_dst), "v"((unsigned long long)gsrc)
               : "memory");
}
// Same, with INST_OFFSET applied to BOTH lds and global address (ISA 08 §4.4).
__device__ __forceinline__ void async_copy16_off(unsigned lds_dst,
                                                 const void* gsrc, int imm) {
  asm volatile("global_load_async_to_lds_b128 %0, %1, off offset:%2"
               :
               : "v"(lds_dst), "v"((unsigned long long)gsrc), "i"(imm)
               : "memory");
}
__device__ __forceinline__ void wait_async0() {
  asm volatile("s_wait_asynccnt 0x0" ::: "memory");
}

// A-matrix fragment (16x32 bf16, row-major LDS, stride `ld` elements, even).
// Contiguous k order (same as B frags here): lanes 0-15 = rows; VGPR q holds K pair starting at
// k = koff + hi*16 + q*2  ->  one aligned b32 load per VGPR.
__device__ __forceinline__ v16bf load_A_frag(const __bf16* base, int ld,
                                             int lane, int koff) {
  const unsigned* p = reinterpret_cast<const unsigned*>(base);
  const int ldu = ld >> 1, ku0 = koff >> 1;
  const int m = lane & 15, hi = (lane >> 4) & 1;
  v8u u;
#pragma unroll
  for (int q = 0; q < 8; ++q) {
    const int ku = ku0 + hi * 8 + q;
    u[q] = p[m * ldu + ku];
  }
  return __builtin_bit_cast(v16bf, u);
}

// Reductions within each 16-lane half of a wave32 (xor masks < 16 stay in half)
__device__ __forceinline__ float half_max(float v) {
  v = fmaxf(v, __shfl_xor(v, 1, 32));
  v = fmaxf(v, __shfl_xor(v, 2, 32));
  v = fmaxf(v, __shfl_xor(v, 4, 32));
  v = fmaxf(v, __shfl_xor(v, 8, 32));
  return v;
}
__device__ __forceinline__ float half_sum(float v) {
  v += __shfl_xor(v, 1, 32);
  v += __shfl_xor(v, 2, 32);
  v += __shfl_xor(v, 4, 32);
  v += __shfl_xor(v, 8, 32);
  return v;
}

// ---------------------------------------------------------------------------
// Kernel 1: QKV projection.  X(16384x768 f32) * Wqkv(768x2304 f32) -> bf16
// scatter into Q/K/V scratch (B,H,N,DH); Q pre-scaled by DH^-0.5.
// 256 threads = 8 waves; 128x64 block tile; wave = 16 rows x 64 cols
// (4 WMMAs per A-fragment -> 4x A reuse, B-frags shared by all waves).
// ---------------------------------------------------------------------------
__global__ __launch_bounds__(256) void qkv_gemm_kernel(
    const float* __restrict__ x, const float* __restrict__ wqkv,
    u16* __restrict__ qsp, u16* __restrict__ ksp, u16* __restrict__ vsp) {
  __bf16* Qs = reinterpret_cast<__bf16*>(qsp);
  __bf16* Ks = reinterpret_cast<__bf16*>(ksp);
  __bf16* Vs = reinterpret_cast<__bf16*>(vsp);

  __shared__ __attribute__((aligned(32))) __bf16 As[128 * 36];  // padded
  __shared__ __attribute__((aligned(32))) __bf16 Bt[64 * 32];   // transposed

  const int tid  = threadIdx.x;
  const int row0 = blockIdx.y * 128;
  const int col0 = blockIdx.x * 64;
  const int lane = tid & 31, w = tid >> 5;
  const int m0 = w * 16;
  const int hi = (lane >> 4) & 1, ln = lane & 15;

  v8f acc[4] = {};
  for (int kk = 0; kk < CDIM; kk += 32) {
    if (kk + 32 < CDIM) {  // prefetch next tiles (global_prefetch_b8)
      __builtin_prefetch(x + (size_t)(row0 + (tid & 127)) * CDIM + kk + 32, 0, 3);
      __builtin_prefetch(
          wqkv + (size_t)(kk + 32 + (tid & 31)) * (3 * CDIM) + col0, 0, 3);
    }
    __syncthreads();
    // A tile: f32 -> bf16, float4 load / v4bf (8B) store. 128x32 elems.
    for (int e4 = tid; e4 < 1024; e4 += 256) {
      const int r = e4 >> 3, c = (e4 & 7) * 4;
      const float4 f =
          *reinterpret_cast<const float4*>(&x[(size_t)(row0 + r) * CDIM + kk + c]);
      v4bf pk;
      pk[0] = (__bf16)f.x; pk[1] = (__bf16)f.y;
      pk[2] = (__bf16)f.z; pk[3] = (__bf16)f.w;
      *reinterpret_cast<v4bf*>(&As[r * 36 + c]) = pk;
    }
    // B tile: float4 row load, transposed bf16 stores.
    for (int e4 = tid; e4 < 512; e4 += 256) {
      const int r = e4 >> 4, c = (e4 & 15) * 4;
      const float4 f = *reinterpret_cast<const float4*>(
          &wqkv[(size_t)(kk + r) * (3 * CDIM) + col0 + c]);
      Bt[(c + 0) * 32 + r] = (__bf16)f.x;
      Bt[(c + 1) * 32 + r] = (__bf16)f.y;
      Bt[(c + 2) * 32 + r] = (__bf16)f.z;
      Bt[(c + 3) * 32 + r] = (__bf16)f.w;
    }
    __syncthreads();
    const v16bf af = load_A_frag(&As[m0 * 36], 36, lane, 0);
#pragma unroll
    for (int t = 0; t < 4; ++t) {
      const v16bf bt =
          *reinterpret_cast<const v16bf*>(&Bt[(t * 16 + ln) * 32 + hi * 16]);
      acc[t] = wmma_bf16(af, bt, acc[t]);
    }
  }

  // Epilogue: q/k/v class is uniform per 16-col tile (16-aligned tiles never
  // cross the 64-aligned head or 768-aligned q/k/v boundaries).
#pragma unroll
  for (int t = 0; t < 4; ++t) {
    const int cb = col0 + t * 16;
    __bf16* dst;
    float scale = 1.0f;
    int cc = cb;
    if (cb < CDIM) {
      dst = Qs; scale = CSCALE;
    } else if (cb < 2 * CDIM) {
      dst = Ks; cc = cb - CDIM;
    } else {
      dst = Vs; cc = cb - 2 * CDIM;
    }
    const int hh = cc >> 6;            // lane-uniform head index
    const int d0 = (cc & 63) + ln;     // dh coordinate for this lane
#pragma unroll
    for (int r = 0; r < 8; ++r) {
      const int grow = row0 + m0 + r + hi * 8;
      const int bb = grow >> 10, nn = grow & (CN - 1);
      dst[(((size_t)bb * CH + hh) * CN + nn) * CDH + d0] =
          (__bf16)(acc[t][r] * scale);
    }
  }
}

// ---------------------------------------------------------------------------
// Kernel 2: permuted-Q flash attention per (b, h, 128-query-row tile).
// 8 waves x 16 query rows; online softmax over 64-key chunks.
// Q gather and K chunks staged with GLOBAL_LOAD_ASYNC_TO_LDS_B128 (ASYNCcnt);
// V staged transposed (sync) so its B-fragments are contiguous 32B LDS loads.
// ---------------------------------------------------------------------------
__global__ __launch_bounds__(256) void attn_kernel(
    const u16* __restrict__ qsp, const u16* __restrict__ ksp,
    const u16* __restrict__ vsp, const int* __restrict__ perm,
    u16* __restrict__ obp) {
  const __bf16* Qs = reinterpret_cast<const __bf16*>(qsp);
  const __bf16* Ks = reinterpret_cast<const __bf16*>(ksp);
  const __bf16* Vs = reinterpret_cast<const __bf16*>(vsp);
  __bf16* Ob = reinterpret_cast<__bf16*>(obp);

  __shared__ __attribute__((aligned(32))) __bf16 Qlds[128 * 72];   // 144B rows
  __shared__ __attribute__((aligned(32))) __bf16 Klds[64 * 64];    // [j][d]
  __shared__ __attribute__((aligned(32))) __bf16 Vt[64 * 64];      // [d][j]
  __shared__ __attribute__((aligned(32))) __bf16 Plds[8][16 * 68]; // per-wave

  const int tid = threadIdx.x, lane = tid & 31, w = tid >> 5;
  const int hi = (lane >> 4) & 1, ln = lane & 15;
  const int b = blockIdx.z, h = blockIdx.y, q0 = blockIdx.x * 128;
  const size_t base = ((size_t)b * CH + h) * CN * CDH;

  // Stage permuted Q rows (q'[i] = q[perm[i]]): async 16B gather copies
  // directly into padded LDS rows (per-lane LDS dest addresses).
  const unsigned qlds0 = lds_off(&Qlds[0]);
  for (int task = tid; task < 512; task += 256) {   // 128 rows x 4 x 16B
    const int row = task >> 2, d8 = task & 3;
    const int src = perm[q0 + row];
    async_copy16(qlds0 + row * 144 + d8 * 16,
                 Qs + base + (size_t)src * CDH + d8 * 8);
  }
  wait_async0();
  __syncthreads();
  const v16bf qf0 = load_A_frag(&Qlds[w * 16 * 72], 72, lane, 0);
  const v16bf qf1 = load_A_frag(&Qlds[w * 16 * 72], 72, lane, 32);

  float mrow[8], lrow[8];
  v8f oacc[4] = {};
#pragma unroll
  for (int r = 0; r < 8; ++r) { mrow[r] = -3.0e38f; lrow[r] = 0.0f; }

  const unsigned klds0 = lds_off(&Klds[0]);
  for (int kc = 0; kc < CN; kc += 64) {
    if (kc + 64 < CN) {  // prefetch next K/V chunk
      __builtin_prefetch(Ks + base + (size_t)(kc + 64) * CDH + tid * 16, 0, 3);
      __builtin_prefetch(Vs + base + (size_t)(kc + 64) * CDH + tid * 16, 0, 3);
    }
    __syncthreads();
    // K chunk: two async b128 copies per lane; the immediate offset applies
    // to both the LDS and global side (contiguous layout on both).
    {
      const void* gk = Ks + base + (size_t)kc * CDH + tid * 8;
      async_copy16(klds0 + tid * 16, gk);
      async_copy16_off(klds0 + tid * 16, gk, 4096);
    }
    // V chunk: transpose into Vt[d][j] (global reads stay coalesced).
    for (int e2 = tid; e2 < 2048; e2 += 256) {
      const int j = e2 >> 5, d2 = e2 & 31;
      const v2bf pv = reinterpret_cast<const v2bf*>(
          Vs + base + (size_t)(kc + j) * CDH)[d2];
      Vt[(2 * d2 + 0) * 64 + j] = pv[0];
      Vt[(2 * d2 + 1) * 64 + j] = pv[1];
    }
    wait_async0();
    __syncthreads();

    // S = Q' K^T (Q pre-scaled); K fragments are contiguous 32B LDS loads.
    v8f s[4];
#pragma unroll
    for (int c = 0; c < 4; ++c) {
      v8f z = {};
      const v16bf kb0 =
          *reinterpret_cast<const v16bf*>(&Klds[(c * 16 + ln) * 64 + hi * 16]);
      const v16bf kb1 = *reinterpret_cast<const v16bf*>(
          &Klds[(c * 16 + ln) * 64 + 32 + hi * 16]);
      z = wmma_bf16(qf0, kb0, z);
      z = wmma_bf16(qf1, kb1, z);
      s[c] = z;
    }

    // Online softmax row stats (row m = r + hi*8 lives in one 16-lane half).
    float mnew[8], alpha[8], psum[8];
#pragma unroll
    for (int r = 0; r < 8; ++r) {
      float v = fmaxf(fmaxf(s[0][r], s[1][r]), fmaxf(s[2][r], s[3][r]));
      v = half_max(v);
      mnew[r]  = fmaxf(mrow[r], v);
      alpha[r] = __expf(mrow[r] - mnew[r]);
      psum[r]  = 0.0f;
    }
#pragma unroll
    for (int c = 0; c < 4; ++c) {
#pragma unroll
      for (int r = 0; r < 8; ++r) {
        const float p = __expf(s[c][r] - mnew[r]);
        psum[r] += p;
        Plds[w][(r + hi * 8) * 68 + c * 16 + ln] = (__bf16)p;  // C->A via LDS
      }
    }
#pragma unroll
    for (int r = 0; r < 8; ++r) {
      lrow[r] = lrow[r] * alpha[r] + half_sum(psum[r]);
      mrow[r] = mnew[r];
#pragma unroll
      for (int c = 0; c < 4; ++c) oacc[c][r] = oacc[c][r] * alpha[r];
    }

    // O += P * V. P frags: b32 loads; V frags: contiguous 32B loads from Vt.
    const v16bf pf0 = load_A_frag(&Plds[w][0], 68, lane, 0);
    const v16bf pf1 = load_A_frag(&Plds[w][0], 68, lane, 32);
#pragma unroll
    for (int c = 0; c < 4; ++c) {
      const int d = c * 16 + ln;
      const v16bf vb0 = *reinterpret_cast<const v16bf*>(&Vt[d * 64 + hi * 16]);
      const v16bf vb1 =
          *reinterpret_cast<const v16bf*>(&Vt[d * 64 + 32 + hi * 16]);
      oacc[c] = wmma_bf16(pf0, vb0, oacc[c]);
      oacc[c] = wmma_bf16(pf1, vb1, oacc[c]);
    }
  }

  float invl[8];
#pragma unroll
  for (int r = 0; r < 8; ++r) invl[r] = 1.0f / lrow[r];
#pragma unroll
  for (int c = 0; c < 4; ++c) {
#pragma unroll
    for (int r = 0; r < 8; ++r) {
      const int qrow = q0 + w * 16 + r + hi * 8;
      Ob[((size_t)b * CN + qrow) * CDIM + h * CDH + c * 16 + ln] =
          (__bf16)(oacc[c][r] * invl[r]);
    }
  }
}

// ---------------------------------------------------------------------------
// Kernel 3: output projection. O(16384x768 bf16) * Wout(768x768) + b -> f32
// 128x64 block tile, wave = 16 rows x 64 cols (4 WMMAs per A-fragment).
// ---------------------------------------------------------------------------
__global__ __launch_bounds__(256) void out_gemm_kernel(
    const u16* __restrict__ obp, const float* __restrict__ wout,
    const float* __restrict__ bout, float* __restrict__ out) {
  const __bf16* Ob = reinterpret_cast<const __bf16*>(obp);

  __shared__ __attribute__((aligned(32))) __bf16 As[128 * 36];
  __shared__ __attribute__((aligned(32))) __bf16 Bt[64 * 32];  // transposed

  const int tid  = threadIdx.x;
  const int row0 = blockIdx.y * 128;
  const int col0 = blockIdx.x * 64;
  const int lane = tid & 31, w = tid >> 5;
  const int m0 = w * 16;
  const int hi = (lane >> 4) & 1, ln = lane & 15;

  v8f acc[4] = {};
  for (int kk = 0; kk < CDIM; kk += 32) {
    if (kk + 32 < CDIM) {
      __builtin_prefetch(Ob + (size_t)(row0 + (tid & 127)) * CDIM + kk + 32, 0, 3);
      __builtin_prefetch(wout + (size_t)(kk + 32 + (tid & 31)) * CDIM + col0, 0, 3);
    }
    __syncthreads();
    // A tile: bf16 row copy via uint2 (8B) moves. 128x32 elems.
    for (int e2 = tid; e2 < 1024; e2 += 256) {
      const int r = e2 >> 3, c = (e2 & 7) * 4;
      *reinterpret_cast<uint2*>(&As[r * 36 + c]) =
          *reinterpret_cast<const uint2*>(&Ob[(size_t)(row0 + r) * CDIM + kk + c]);
    }
    // B tile: float4 row load, transposed bf16 stores.
    for (int e4 = tid; e4 < 512; e4 += 256) {
      const int r = e4 >> 4, c = (e4 & 15) * 4;
      const float4 f = *reinterpret_cast<const float4*>(
          &wout[(size_t)(kk + r) * CDIM + col0 + c]);
      Bt[(c + 0) * 32 + r] = (__bf16)f.x;
      Bt[(c + 1) * 32 + r] = (__bf16)f.y;
      Bt[(c + 2) * 32 + r] = (__bf16)f.z;
      Bt[(c + 3) * 32 + r] = (__bf16)f.w;
    }
    __syncthreads();
    const v16bf af = load_A_frag(&As[m0 * 36], 36, lane, 0);
#pragma unroll
    for (int t = 0; t < 4; ++t) {
      const v16bf bt =
          *reinterpret_cast<const v16bf*>(&Bt[(t * 16 + ln) * 32 + hi * 16]);
      acc[t] = wmma_bf16(af, bt, acc[t]);
    }
  }

#pragma unroll
  for (int t = 0; t < 4; ++t) {
    const int col = col0 + t * 16 + ln;
    const float bias = bout[col];
#pragma unroll
    for (int r = 0; r < 8; ++r) {
      const int grow = row0 + m0 + r + hi * 8;
      out[(size_t)grow * CDIM + col] = acc[t][r] + bias;
    }
  }
}

// ---------------------------------------------------------------------------
// Launcher: 3 kernels on `stream`, scratch carved out of d_ws (bf16, 96 MB).
// ---------------------------------------------------------------------------
extern "C" void kernel_launch(void* const* d_in, const int* in_sizes, int n_in,
                              void* d_out, int out_size, void* d_ws,
                              size_t ws_size, hipStream_t stream) {
  const float* x    = (const float*)d_in[0];
  const float* wqkv = (const float*)d_in[1];
  const float* wout = (const float*)d_in[2];
  const float* bout = (const float*)d_in[3];
  const int*   perm = (const int*)d_in[4];
  float*       out  = (float*)d_out;

  const size_t elems = (size_t)CB * CH * CN * CDH;  // 12,582,912 (== B*N*DIM)
  u16* qs = (u16*)d_ws;
  u16* ks = qs + elems;
  u16* vs = ks + elems;
  u16* ob = vs + elems;

  qkv_gemm_kernel<<<dim3((3 * CDIM) / 64, (CB * CN) / 128), 256, 0, stream>>>(
      x, wqkv, qs, ks, vs);
  attn_kernel<<<dim3(CN / 128, CH, CB), 256, 0, stream>>>(qs, ks, vs, perm, ob);
  out_gemm_kernel<<<dim3(CDIM / 64, (CB * CN) / 128), 256, 0, stream>>>(
      ob, wout, bout, out);
}
